// CINTransform_72378788872991
// MI455X (gfx1250) — matrix operand order repaired and checked by reading, not compiled, not run on hardware
//
#include <hip/hip_runtime.h>

// ============================================================================
// out[b,h,f] = sum_{i,j} in1[b,i,f] * W[h,i,j] * in2[b,j,f]
// B=2048, H=16, D1=D2=64, F=128, all fp32.
//
// Strategy (MI455X / gfx1250, wave32, WMMA):
//   per (b,h):  T[i,f] = W_h[i,:] x in2[b][:,f]   (64x64 @ 64x128 GEMM)
//               out[b,h,f] = sum_i in1[b,i,f] * T[i,f]
//   GEMM runs on v_wmma_f32_16x16x32_f16 (f16 in, f32 accumulate).
//   W is converted to f16 A-fragments in LDS once per workgroup (128 KB),
//   reused by NB batches x 16 heads. in2[b] lives in B-frag registers for
//   all 16 heads; in1[b] lives in fp32 registers in C-tile layout.
// ============================================================================

typedef __attribute__((ext_vector_type(16))) _Float16 v16h;
typedef __attribute__((ext_vector_type(8)))  float    v8f;

#define B_TOT    2048
#define H_DIM    16
#define D_DIM    64
#define F_DIM    128
#define NB       4            // batches per workgroup -> 512 workgroups
#define NTHREADS 256          // 8 waves; wave w owns f-tile [16w, 16w+16)
#define NFRAGS   (H_DIM * 2 * 4)       // h * kt * mt = 128 fragments
#define LDS_BYTES (NFRAGS * 1024)      // 131072 B = 128 KB

struct alignas(16) U4 { unsigned int x, y, z, w; };

// A-fragment (16-bit, 16x32) K position for half-index idx in lane half 'hi':
//   idx 0..7 -> K = idx, idx 8..15 -> K = idx+8; lanes 16..31 add +8.
__device__ __forceinline__ int a_kmap(int idx, int hi) {
    return (idx < 8 ? idx : idx + 8) + hi * 8;
}

__global__ __launch_bounds__(NTHREADS)
void trilinear_wmma_kernel(const float* __restrict__ in1,
                           const float* __restrict__ in2,
                           const float* __restrict__ w,
                           float* __restrict__ out)
{
    extern __shared__ char lds[];   // 128 KB of f16 A-fragments of W

    const int tid   = threadIdx.x;
    const int wave  = tid >> 5;       // 0..7  -> f-tile
    const int lane  = tid & 31;
    const int laneN = lane & 15;      // column within 16-wide tile
    const int hi    = lane >> 4;      // which half of the wave

    // ---- Stage W (fp32, global) -> LDS as f16 WMMA A-fragments -------------
    // fragId = ((h*2 + kt)*4 + mt); per-lane 16 halfs stored as two 16B
    // chunks: frag*1024 + chunk*512 + lane*16  (conflict-free b128 reads).
    for (int k = 0; k < 16; ++k) {
        const int r    = tid + NTHREADS * k;   // 0..4095 lane-rows
        const int frag = r >> 5;
        const int lr   = r & 31;
        const int mt   = frag & 3;
        const int kt   = (frag >> 2) & 1;
        const int h    = frag >> 3;
        const int lrN  = lr & 15, lrH = lr >> 4;
        const int i    = mt * 16 + lrN;        // A-matrix row (M = i)
        _Float16 tmp[16];
        #pragma unroll
        for (int idx = 0; idx < 16; ++idx) {
            const int j = kt * 32 + a_kmap(idx, lrH);   // A-matrix K = j
            tmp[idx] = (_Float16)w[(h * D_DIM + i) * D_DIM + j];
        }
        char* dst = lds + frag * 1024 + lr * 16;
        *(U4*)(dst)       = *(const U4*)&tmp[0];
        *(U4*)(dst + 512) = *(const U4*)&tmp[8];
    }
    __syncthreads();

    const int f0 = wave * 16;
    const int b0 = blockIdx.x * NB;

    for (int bt = 0; bt < NB; ++bt) {
        const int b = b0 + bt;
        const float* in1b = in1 + (size_t)b * D_DIM * F_DIM;
        const float* in2b = in2 + (size_t)b * D_DIM * F_DIM;

        // Prefetch next batch (global_prefetch_b8); 256 threads x 128B
        // covers each 32 KB input slab.
        if (bt + 1 < NB) {
            __builtin_prefetch(in1b + D_DIM * F_DIM + tid * 32, 0, 0);
            __builtin_prefetch(in2b + D_DIM * F_DIM + tid * 32, 0, 0);
        }

        // ---- B-fragments: in2[b][j, f0+laneN] -> f16, same K layout as A:
        // K = a_kmap(idx, hi) (idx 0..7 -> idx, 8..15 -> idx+8; lanes 16-31 +8).
        v16h bfrag[2];
        #pragma unroll
        for (int kt = 0; kt < 2; ++kt) {
            #pragma unroll
            for (int idx = 0; idx < 16; ++idx) {
                const int j = kt * 32 + a_kmap(idx, hi);
                bfrag[kt][idx] = (_Float16)in2b[j * F_DIM + f0 + laneN];
            }
        }

        // ---- in1[b] in C/D-tile layout (fp32, reused by all 16 heads):
        // acc element vg of tile mt sits at (i = mt*16 + vg + 8*hi, f = f0+laneN)
        float a1[32];
        #pragma unroll
        for (int mt = 0; mt < 4; ++mt)
            #pragma unroll
            for (int vg = 0; vg < 8; ++vg) {
                const int i = mt * 16 + vg + hi * 8;
                a1[mt * 8 + vg] = in1b[i * F_DIM + f0 + laneN];
            }

        // ---- Head loop: 8 WMMAs each, then fused elementwise reduce --------
        for (int h = 0; h < H_DIM; ++h) {
            v8f acc[4] = {};
            #pragma unroll
            for (int kt = 0; kt < 2; ++kt) {
                #pragma unroll
                for (int mt = 0; mt < 4; ++mt) {
                    const char* src =
                        lds + (((h * 2 + kt) * 4 + mt) * 1024) + lane * 16;
                    union { v16h v; U4 q[2]; } af;
                    af.q[0] = *(const U4*)(src);
                    af.q[1] = *(const U4*)(src + 512);
                    acc[mt] = __builtin_amdgcn_wmma_f32_16x16x32_f16(
                        false, af.v, false, bfrag[kt],
                        (short)0, acc[mt], false, false);
                }
            }
            // out[b,h,f] = sum_i in1[b,i,f] * T[i,f]
            float s = 0.f;
            #pragma unroll
            for (int mt = 0; mt < 4; ++mt)
                #pragma unroll
                for (int vg = 0; vg < 8; ++vg)
                    s = fmaf(a1[mt * 8 + vg], acc[mt][vg], s);
            s += __shfl_xor(s, 16, 32);   // combine the two wave halves
            if (hi == 0)
                out[((size_t)b * H_DIM + h) * F_DIM + f0 + laneN] = s;
        }
    }
}

extern "C" void kernel_launch(void* const* d_in, const int* in_sizes, int n_in,
                              void* d_out, int out_size, void* d_ws, size_t ws_size,
                              hipStream_t stream) {
    (void)in_sizes; (void)n_in; (void)d_ws; (void)ws_size; (void)out_size;
    const float* in1 = (const float*)d_in[0];   // [B, D1, F]
    const float* in2 = (const float*)d_in[1];   // [B, D2, F]
    const float* w   = (const float*)d_in[2];   // [H, D1, D2]
    float* out = (float*)d_out;                 // [B, H, F]
    dim3 grid(B_TOT / NB), block(NTHREADS);
    trilinear_wmma_kernel<<<grid, block, LDS_BYTES, stream>>>(in1, in2, w, out);
}
